// MQAAttention_11192684773438
// MI455X (gfx1250) — matrix operand order repaired and checked by reading, not compiled, not run on hardware
//
#include <hip/hip_runtime.h>

#define DIM    1024
#define NHEADS 16
#define HDIM   64
#define SEQ    2048
#define NB     2
#define ROWS   (NB * SEQ)  // 4096

typedef __attribute__((ext_vector_type(16))) _Float16 v16h;
typedef __attribute__((ext_vector_type(8)))  _Float16 h8;
typedef __attribute__((ext_vector_type(8)))  float    v8f;

union Frag { v16h v; h8 h[2]; };

static __device__ __forceinline__ v8f wmma16(v16h a, v16h b, v8f c) {
  // (neg_a, A, neg_b, B, c_mod, C, reuse_a, reuse_b)
  return __builtin_amdgcn_wmma_f32_16x16x32_f16(false, a, false, b, (short)0, c,
                                                false, false);
}

static __device__ __forceinline__ v8f zero8() {
  v8f z = {0.f, 0.f, 0.f, 0.f, 0.f, 0.f, 0.f, 0.f};
  return z;
}

static __device__ __forceinline__ float redmax16(float x) {
  x = fmaxf(x, __shfl_xor(x, 1));
  x = fmaxf(x, __shfl_xor(x, 2));
  x = fmaxf(x, __shfl_xor(x, 4));
  x = fmaxf(x, __shfl_xor(x, 8));
  return x;
}
static __device__ __forceinline__ float redsum16(float x) {
  x += __shfl_xor(x, 1);
  x += __shfl_xor(x, 2);
  x += __shfl_xor(x, 4);
  x += __shfl_xor(x, 8);
  return x;
}

// ---------------------------------------------------------- async copy plumbing
#if defined(__AMDGCN__) && __has_builtin(__builtin_amdgcn_global_load_async_to_lds_b128) && \
    __has_builtin(__builtin_amdgcn_s_wait_asynccnt)
#define ASYNC_COPY 1
#else
#define ASYNC_COPY 0
#endif

#if ASYNC_COPY
typedef int v4i __attribute__((vector_size(16)));
typedef __attribute__((address_space(1))) v4i* g4p;  // global int4*
typedef __attribute__((address_space(3))) v4i* l4p;  // LDS int4*
#endif

static __device__ __forceinline__ void cp16_to_lds(const _Float16* g, _Float16* l) {
#if ASYNC_COPY
  __builtin_amdgcn_global_load_async_to_lds_b128((g4p)(void*)g, (l4p)(void*)l, 0, 0);
#else
  *(h8*)l = *(const h8*)g;  // sync fallback: global_load_b128 + ds_store_b128
#endif
}

#if ASYNC_COPY
#define ASYNC_WAIT(n) __builtin_amdgcn_s_wait_asynccnt(n)
#else
#define ASYNC_WAIT(n)
#endif

// ---------------------------------------------------------------- conversions
__global__ void cvt_f32_f16(const float* __restrict__ in,
                            _Float16* __restrict__ out, int n) {
  int i = blockIdx.x * 256 + threadIdx.x;
  if (i < n) out[i] = (_Float16)in[i];
}

// ------------------------------------------------------------------- GEMM
// C[M x N] = A[M x K] @ W[N x K]^T ; A,W fp16, C fp32.
// Block: 256 threads = 8 waves. Block tile 128x64, K-step 32, double-buffered
// LDS staging through the gfx1250 async copy engine (ASYNCcnt).
__global__ void __launch_bounds__(256)
gemm_f16(const _Float16* __restrict__ A, const _Float16* __restrict__ W,
         float* __restrict__ C, int N, int K) {
  __shared__ __align__(16) _Float16 Asm[2][128 * 32];  // 8 KB each
  __shared__ __align__(16) _Float16 Bsm[2][64 * 32];   // 4 KB each

  const int tid  = threadIdx.x;
  const int wave = tid >> 5;
  const int lane = tid & 31;
  const int m    = lane & 15;
  const int half = lane >> 4;
  const int m0 = blockIdx.x * 128;
  const int n0 = blockIdx.y * 64;

  // copy assignment: row = tid>>2 in [0,64), col = (tid&3)*8 ; 3 x b128 per thread
  const int crow = tid >> 2;
  const int ccol = (tid & 3) * 8;
  const _Float16* ga0 = A + (size_t)(m0 + crow) * K + ccol;
  const _Float16* ga1 = A + (size_t)(m0 + 64 + crow) * K + ccol;
  const _Float16* gb  = W + (size_t)(n0 + crow) * K + ccol;

  v8f acc[4] = {zero8(), zero8(), zero8(), zero8()};
  const int nk = K / 32;

  // prologue: stage tile 0 into buffer 0
  cp16_to_lds(ga0, &Asm[0][crow * 32 + ccol]);
  cp16_to_lds(ga1, &Asm[0][(64 + crow) * 32 + ccol]);
  cp16_to_lds(gb,  &Bsm[0][crow * 32 + ccol]);

  for (int kk = 0; kk < nk; ++kk) {
    const int buf = kk & 1;
    if (kk + 1 < nk) {
      const int k1 = (kk + 1) * 32;
      cp16_to_lds(ga0 + k1, &Asm[buf ^ 1][crow * 32 + ccol]);
      cp16_to_lds(ga1 + k1, &Asm[buf ^ 1][(64 + crow) * 32 + ccol]);
      cp16_to_lds(gb + k1,  &Bsm[buf ^ 1][crow * 32 + ccol]);
      ASYNC_WAIT(3);  // copies for tile kk complete (in-order)
    } else {
      ASYNC_WAIT(0);
    }
    __syncthreads();

    // fragments from LDS
    const _Float16* arow = &Asm[buf][(wave * 16 + m) * 32];
    Frag a;
    a.h[0] = *(const h8*)(arow + half * 8);        // K = 0..7 / 8..15
    a.h[1] = *(const h8*)(arow + 16 + half * 8);   // K = 16..23 / 24..31
#pragma unroll
    for (int t = 0; t < 4; ++t) {
      const _Float16* brow = &Bsm[buf][(t * 16 + m) * 32];
      Frag b;
      b.h[0] = *(const h8*)(brow + half * 8);       // K = 0..7 / 8..15
      b.h[1] = *(const h8*)(brow + 16 + half * 8);  // K = 16..23 / 24..31
      acc[t] = wmma16(a.v, b.v, acc[t]);
    }
    __syncthreads();
  }
#pragma unroll
  for (int t = 0; t < 4; ++t)
#pragma unroll
    for (int r = 0; r < 8; ++r)
      C[(size_t)(m0 + wave * 16 + r + half * 8) * N + n0 + t * 16 + m] = acc[t][r];
}

// ------------------------------------------------------------------- RoPE
__global__ void rope_q(const float* __restrict__ qf, _Float16* __restrict__ qh) {
  int idx = blockIdx.x * 256 + threadIdx.x;  // (b,s,h,i) : 2*2048*16*32
  int i = idx & 31;
  int h = (idx >> 5) & 15;
  int s = (idx >> 9) & 2047;
  int b = idx >> 20;
  float freq = __expf(-0.2878231366f * (float)i);  // theta^(-2i/64)
  float sn, cs;
  __sincosf((float)s * freq, &sn, &cs);
  const float* src = qf + (size_t)(b * SEQ + s) * DIM + h * HDIM + 2 * i;
  float x1 = src[0], x2 = src[1];
  _Float16* dst = qh + ((size_t)(b * NHEADS + h) * SEQ + s) * HDIM + 2 * i;
  dst[0] = (_Float16)((x1 * cs - x2 * sn) * 0.125f);  // fold 1/sqrt(64)
  dst[1] = (_Float16)((x1 * sn + x2 * cs) * 0.125f);
}

__global__ void rope_k(const float* __restrict__ kf, _Float16* __restrict__ kh) {
  int idx = blockIdx.x * 256 + threadIdx.x;  // (b,s,i) : 2*2048*32
  int i = idx & 31;
  int s = (idx >> 5) & 2047;
  int b = idx >> 16;
  float freq = __expf(-0.2878231366f * (float)i);
  float sn, cs;
  __sincosf((float)s * freq, &sn, &cs);
  const float* src = kf + (size_t)(b * SEQ + s) * HDIM + 2 * i;
  float x1 = src[0], x2 = src[1];
  _Float16* dst = kh + (size_t)(b * SEQ + s) * HDIM + 2 * i;
  dst[0] = (_Float16)(x1 * cs - x2 * sn);
  dst[1] = (_Float16)(x1 * sn + x2 * cs);
}

// V (B,S,64) fp32 -> Vt (B,64,S) fp16 so P@V B-fragments are contiguous.
__global__ void cvt_vt(const float* __restrict__ vf, _Float16* __restrict__ vt) {
  int idx = blockIdx.x * 256 + threadIdx.x;  // (b,s,d) : 2*2048*64
  int d = idx & 63;
  int s = (idx >> 6) & 2047;
  int b = idx >> 17;
  vt[((size_t)(b * HDIM + d)) * SEQ + s] = (_Float16)vf[(size_t)(b * SEQ + s) * HDIM + d];
}

// ------------------------------------------------------------ flash attention
// One wave per (b, h, 16-query tile). Online-softmax over 32-key chunks.
__global__ void __launch_bounds__(32)
attn_kernel(const _Float16* __restrict__ qh, const _Float16* __restrict__ kh,
            const _Float16* __restrict__ vt, float* __restrict__ o) {
  __shared__ __align__(16) _Float16 ptile[16 * 32];
  const int lane = threadIdx.x & 31;
  const int m    = lane & 15;
  const int half = lane >> 4;
  const int q0 = blockIdx.x * 16;
  const int h  = blockIdx.y;
  const int b  = blockIdx.z;

  const _Float16* qrow = qh + ((size_t)(b * NHEADS + h) * SEQ + q0 + m) * HDIM;
  Frag aq0, aq1;  // Q rows, contraction chunks d=0..31 and d=32..63
  aq0.h[0] = *(const h8*)(qrow + half * 8);
  aq0.h[1] = *(const h8*)(qrow + 16 + half * 8);
  aq1.h[0] = *(const h8*)(qrow + 32 + half * 8);
  aq1.h[1] = *(const h8*)(qrow + 48 + half * 8);

  float mrun[8], lrun[8];
#pragma unroll
  for (int r = 0; r < 8; ++r) { mrun[r] = -1e30f; lrun[r] = 0.f; }
  v8f accO[4] = {zero8(), zero8(), zero8(), zero8()};

  const int kend = q0 + 16;
  for (int kv0 = 0; kv0 < kend; kv0 += 32) {
    v8f sacc[2] = {zero8(), zero8()};
#pragma unroll
    for (int nt = 0; nt < 2; ++nt) {
      const _Float16* krow = kh + ((size_t)b * SEQ + kv0 + nt * 16 + m) * HDIM;
      Frag bk0, bk1;
      bk0.h[0] = *(const h8*)(krow + half * 8);
      bk0.h[1] = *(const h8*)(krow + 16 + half * 8);
      bk1.h[0] = *(const h8*)(krow + 32 + half * 8);
      bk1.h[1] = *(const h8*)(krow + 48 + half * 8);
      sacc[nt] = wmma16(aq0.v, bk0.v, sacc[nt]);
      sacc[nt] = wmma16(aq1.v, bk1.v, sacc[nt]);
    }
    // causal mask + online softmax (element: M = r + 8*half, N = nt*16 + m)
#pragma unroll
    for (int r = 0; r < 8; ++r) {
      const int mq = q0 + r + half * 8;
      float s0 = sacc[0][r];
      float s1 = sacc[1][r];
      if (kv0 + m > mq)      s0 = -1e30f;
      if (kv0 + 16 + m > mq) s1 = -1e30f;
      float cur  = redmax16(fmaxf(s0, s1));
      float mnew = fmaxf(mrun[r], cur);
      float p0 = __expf(s0 - mnew);
      float p1 = __expf(s1 - mnew);
      float rs = redsum16(p0 + p1);
      float corr = __expf(mrun[r] - mnew);
      lrun[r] = lrun[r] * corr + rs;
      mrun[r] = mnew;
#pragma unroll
      for (int t = 0; t < 4; ++t) accO[t][r] = accO[t][r] * corr;
      ptile[(r + half * 8) * 32 + m]      = (_Float16)p0;
      ptile[(r + half * 8) * 32 + 16 + m] = (_Float16)p1;
    }
    __syncthreads();
    Frag ap;  // P as A-matrix fragment (16x32)
    ap.h[0] = *(const h8*)(&ptile[m * 32 + half * 8]);
    ap.h[1] = *(const h8*)(&ptile[m * 32 + 16 + half * 8]);
#pragma unroll
    for (int t = 0; t < 4; ++t) {
      const _Float16* vrow =
          vt + ((size_t)(b * HDIM + t * 16 + m)) * SEQ + kv0 + half * 8;
      Frag bv;
      bv.h[0] = *(const h8*)(vrow);
      bv.h[1] = *(const h8*)(vrow + 16);
      accO[t] = wmma16(ap.v, bv.v, accO[t]);
    }
    __syncthreads();
  }
  // write O as (B, S, H*Dh) fp32, normalized by softmax denom
#pragma unroll
  for (int r = 0; r < 8; ++r) {
    float inv = 1.0f / lrun[r];
    float* orow = o + ((size_t)(b * SEQ + q0 + r + half * 8)) * DIM + h * HDIM;
#pragma unroll
    for (int t = 0; t < 4; ++t) orow[t * 16 + m] = accO[t][r] * inv;
  }
}

// ------------------------------------------------------------------ RMS norm
__global__ void __launch_bounds__(256)
rmsnorm_f16(const float* __restrict__ o, const float* __restrict__ w,
            _Float16* __restrict__ out) {
  __shared__ float red[8];
  const int row = blockIdx.x;
  const float* x = o + (size_t)row * DIM;
  float ss = 0.f;
  for (int j = threadIdx.x; j < DIM; j += 256) { float v = x[j]; ss = fmaf(v, v, ss); }
  ss += __shfl_xor(ss, 1);
  ss += __shfl_xor(ss, 2);
  ss += __shfl_xor(ss, 4);
  ss += __shfl_xor(ss, 8);
  ss += __shfl_xor(ss, 16);
  if ((threadIdx.x & 31) == 0) red[threadIdx.x >> 5] = ss;
  __syncthreads();
  float tot = 0.f;
#pragma unroll
  for (int i = 0; i < 8; ++i) tot += red[i];
  float sc = rsqrtf(tot * (1.0f / DIM) + 1e-6f);
  for (int j = threadIdx.x; j < DIM; j += 256)
    out[(size_t)row * DIM + j] = (_Float16)(x[j] * sc * w[j]);
}

// ------------------------------------------------------------------- launch
extern "C" void kernel_launch(void* const* d_in, const int* in_sizes, int n_in,
                              void* d_out, int out_size, void* d_ws, size_t ws_size,
                              hipStream_t stream) {
  (void)in_sizes; (void)n_in; (void)out_size; (void)ws_size;
  const float* x      = (const float*)d_in[0];
  const float* Wq     = (const float*)d_in[1];
  const float* Wk     = (const float*)d_in[2];
  const float* Wv     = (const float*)d_in[3];
  const float* Wo     = (const float*)d_in[4];
  const float* norm_w = (const float*)d_in[5];
  float* out = (float*)d_out;

  char* base = (char*)d_ws;
  size_t off = 0;
  auto alloc = [&](size_t bytes) {
    size_t o = off;
    off += (bytes + 255) & ~(size_t)255;
    return o;
  };
  _Float16* xh  = (_Float16*)(base + alloc((size_t)ROWS * DIM * 2));   // 8 MB
  _Float16* wqh = (_Float16*)(base + alloc((size_t)DIM * DIM * 2));    // 2 MB
  _Float16* wkh = (_Float16*)(base + alloc((size_t)HDIM * DIM * 2));
  _Float16* wvh = (_Float16*)(base + alloc((size_t)HDIM * DIM * 2));
  _Float16* woh = (_Float16*)(base + alloc((size_t)DIM * DIM * 2));    // 2 MB
  float*    qf  = (float*)(base + alloc((size_t)ROWS * DIM * 4));      // 16 MB (reused as attn O)
  float*    kf  = (float*)(base + alloc((size_t)ROWS * HDIM * 4));     // 1 MB
  float*    vf  = (float*)(base + alloc((size_t)ROWS * HDIM * 4));     // 1 MB
  _Float16* qhh = (_Float16*)(base + alloc((size_t)NB * NHEADS * SEQ * HDIM * 2)); // 8 MB
  _Float16* khh = (_Float16*)(base + alloc((size_t)NB * SEQ * HDIM * 2));
  _Float16* vth = (_Float16*)(base + alloc((size_t)NB * HDIM * SEQ * 2));
  float*    ofp = qf;  // reuse: qf consumed by rope_q before attention writes O
  _Float16* nrm = xh;  // reuse: xh consumed by projection GEMMs

  // 1) fp32 -> fp16
  cvt_f32_f16<<<(ROWS * DIM + 255) / 256, 256, 0, stream>>>(x, xh, ROWS * DIM);
  cvt_f32_f16<<<(DIM * DIM + 255) / 256, 256, 0, stream>>>(Wq, wqh, DIM * DIM);
  cvt_f32_f16<<<(HDIM * DIM + 255) / 256, 256, 0, stream>>>(Wk, wkh, HDIM * DIM);
  cvt_f32_f16<<<(HDIM * DIM + 255) / 256, 256, 0, stream>>>(Wv, wvh, HDIM * DIM);
  cvt_f32_f16<<<(DIM * DIM + 255) / 256, 256, 0, stream>>>(Wo, woh, DIM * DIM);

  // 2) projections: q = x@Wq^T (4096x1024), k/v = x@Wk^T / x@Wv^T (4096x64)
  gemm_f16<<<dim3(ROWS / 128, DIM / 64), 256, 0, stream>>>(xh, wqh, qf, DIM, DIM);
  gemm_f16<<<dim3(ROWS / 128, 1), 256, 0, stream>>>(xh, wkh, kf, HDIM, DIM);
  gemm_f16<<<dim3(ROWS / 128, 1), 256, 0, stream>>>(xh, wvh, vf, HDIM, DIM);

  // 3) RoPE (+1/sqrt(Dh) folded into Q), V transpose
  rope_q<<<(NB * SEQ * NHEADS * 32) / 256, 256, 0, stream>>>(qf, qhh);
  rope_k<<<(NB * SEQ * 32) / 256, 256, 0, stream>>>(kf, khh);
  cvt_vt<<<(NB * SEQ * HDIM) / 256, 256, 0, stream>>>(vf, vth);

  // 4) causal flash attention -> O (B,S,1024) fp32
  attn_kernel<<<dim3(SEQ / 16, NHEADS, NB), 32, 0, stream>>>(qhh, khh, vth, ofp);

  // 5) RMS norm -> fp16
  rmsnorm_f16<<<ROWS, 256, 0, stream>>>(ofp, norm_w, nrm);

  // 6) final projection: out = normed @ Wo^T
  gemm_f16<<<dim3(ROWS / 128, DIM / 64), 256, 0, stream>>>(nrm, woh, out, DIM, DIM);
}
